// TPRencoder_LSTM_41351945126553
// MI455X (gfx1250) — compile-verified, matrix operand order checked by reading, not run on hardware
//
#include <hip/hip_runtime.h>
#include <hip/hip_bf16.h>

// ---------------- problem constants ----------------
#define BATCH 64
#define SEQ   256
#define INDIM 400
#define INP   416                 // 400 padded to 13*32
#define HID   1024
#define H4    4096
#define KCAT  (INP + HID + 32)    // 1472: [x | T | bias-tile]
#define NS    150
#define NSP   160
#define NR    50
#define NRP   64
#define NBLK  128                 // persistent grid: 128 blocks x 256 thr = 1024 waves

typedef __attribute__((ext_vector_type(16))) __bf16 v16bf;
typedef __attribute__((ext_vector_type(8)))  float  v8f;

// Opaque compiler barrier: forbids hoisting/sinking memory ops across it.
// Used once per scan step so loop-invariant weight loads are NOT hoisted
// out of the 256-iteration loop (which would spill to scratch).
#define OPT_BARRIER() asm volatile("" ::: "memory")

// ===================================================================
// f32 -> bf16 convert with zero padding of rows/cols (grid-stride)
// ===================================================================
__global__ __launch_bounds__(256)
void cvt_pad_kernel(const float* __restrict__ src, unsigned short* __restrict__ dstRaw,
                    int src_rows, int src_cols, int dst_rows, int dst_cols)
{
    __bf16* dst = (__bf16*)dstRaw;
    size_t n = (size_t)dst_rows * dst_cols;
    for (size_t i = (size_t)blockIdx.x * 256 + threadIdx.x; i < n;
         i += (size_t)gridDim.x * 256) {
        int r = (int)(i / dst_cols);
        int c = (int)(i % dst_cols);
        float v = (r < src_rows && c < src_cols) ? src[(size_t)r * src_cols + c] : 0.0f;
        dst[i] = (__bf16)v;
    }
}

// ===================================================================
// Build gate-permuted, bias-augmented combined weight  [4096 x 1472] bf16.
// Permuted row n' = 4*j + g  maps to original stacked-gate row g*1024 + j.
// cols 0..415   : Wih (zero-padded from 400)
// cols 416..1439: Whh
// col  1440     : bih + bhh   (consumed by the constant-1 A column)
// ===================================================================
__global__ __launch_bounds__(256)
void build_wcat_kernel(const float* __restrict__ Wih, const float* __restrict__ Whh,
                       const float* __restrict__ bih, const float* __restrict__ bhh,
                       unsigned short* __restrict__ outRaw)
{
    __bf16* out = (__bf16*)outRaw;
    const size_t n = (size_t)H4 * KCAT;
    for (size_t idx = (size_t)blockIdx.x * 256 + threadIdx.x; idx < n;
         idx += (size_t)gridDim.x * 256) {
        int row = (int)(idx / KCAT);          // permuted row: 4*j + g
        int col = (int)(idx % KCAT);
        int j = row >> 2, g = row & 3;
        int orow = g * HID + j;               // original row in [4H, *] weights
        float v;
        if (col < INP)                 v = (col < INDIM) ? Wih[(size_t)orow * INDIM + col] : 0.0f;
        else if (col < INP + HID)      v = Whh[(size_t)orow * HID + (col - INP)];
        else if (col == INP + HID)     v = bih[orow] + bhh[orow];
        else                           v = 0.0f;
        out[idx] = (__bf16)v;
    }
}

// ===================================================================
// init: zero T (bf16), cF, cR (f32), build [1,0,..,0] column, reset barrier.
// ===================================================================
__global__ __launch_bounds__(256)
void init_state_kernel(unsigned short* __restrict__ TbfRaw,
                       float* __restrict__ cF, float* __restrict__ cR,
                       unsigned short* __restrict__ onesRaw,
                       int* __restrict__ barrier_state)
{
    int i = blockIdx.x * 256 + threadIdx.x;   // 0 .. 65535
    ((__bf16*)TbfRaw)[i] = (__bf16)0.0f;
    cF[i] = 0.0f;
    cR[i] = 0.0f;
    if (blockIdx.x == 0 && threadIdx.x < 32)
        ((__bf16*)onesRaw)[threadIdx.x] = (threadIdx.x == 0) ? (__bf16)1.0f : (__bf16)0.0f;
    if (blockIdx.x == 0 && threadIdx.x < 2)
        barrier_state[threadIdx.x] = 0;       // [0]=arrive counter, [1]=generation
}

// ===================================================================
// device-wide barrier: generation counter + agent-scope atomics.
// Safe: all NBLK blocks are co-resident (small LDS/VGPR footprint).
// ===================================================================
__device__ __forceinline__ void grid_sync(int* cnt, int* gen)
{
    __syncthreads();
    if (threadIdx.x == 0) {
        __threadfence();                      // release this block's global writes
        int g = __hip_atomic_load(gen, __ATOMIC_RELAXED, __HIP_MEMORY_SCOPE_AGENT);
        int prev = __hip_atomic_fetch_add(cnt, 1, __ATOMIC_ACQ_REL, __HIP_MEMORY_SCOPE_AGENT);
        if (prev == NBLK - 1) {
            __hip_atomic_store(cnt, 0, __ATOMIC_RELAXED, __HIP_MEMORY_SCOPE_AGENT);
            __hip_atomic_fetch_add(gen, 1, __ATOMIC_ACQ_REL, __HIP_MEMORY_SCOPE_AGENT);
        } else {
            while (__hip_atomic_load(gen, __ATOMIC_ACQUIRE, __HIP_MEMORY_SCOPE_AGENT) == g)
                __builtin_amdgcn_s_sleep(1);
        }
        __threadfence();                      // acquire other blocks' writes
    }
    __syncthreads();
}

// ===================================================================
// PERSISTENT SCAN KERNEL: the entire 256-step recurrence in one launch.
// Per step:
//  phase 1 (all 1024 waves): gates = [x_t | T | 1] @ Wcat^T via bf16 WMMA
//     (gate columns permuted 4*j+g, biases folded into K), LDS regroup,
//     LSTM activations -> c update, h (bf16).        --- grid_sync ---
//  phase 2 (blocks 0..3):   logits = h @ Wa^T (WMMA -> LDS), softmax,
//     item projections, outer product -> T_t.        --- grid_sync ---
// NOTE: read-only operands are deliberately NOT __restrict__ and each step
// starts with an opaque memory barrier, so the compiler cannot hoist the
// (loop-invariant) weight-fragment loads out of the scan loop — they must
// issue from L2 every step instead of spilling to scratch.
// ===================================================================
__global__ __launch_bounds__(256)
void tpr_scan_kernel(const unsigned short* XbfRaw,
                     unsigned short* TbfRaw,
                     const unsigned short* WcatFRaw,
                     const unsigned short* WcatRRaw,
                     const unsigned short* WaFRaw,
                     const unsigned short* WaRRaw,
                     const unsigned short* onesRaw,
                     float* cF, float* cR,
                     unsigned short* hFRaw, unsigned short* hRRaw,
                     const float* WaFb, const float* WaRb,
                     const float* Fw, const float* Fb,
                     const float* Rw, const float* Rb,
                     float* outT, float* outAF, float* outAR,
                     int* barrier_state)
{
    __shared__ float ltile[8][16][33];        // phase 1: per-wave 16x32 tile
    __shared__ float lgF_s[16][NSP];          // phase 2
    __shared__ float lgR_s[16][NRP];
    __shared__ float itF_s[16][32];
    __shared__ float itR_s[16][32];

    int* bar_cnt = barrier_state;
    int* bar_gen = barrier_state + 1;

    const __bf16* Xbf  = (const __bf16*)XbfRaw;
    const __bf16* Tbf  = (const __bf16*)TbfRaw;
    const __bf16* ones = (const __bf16*)onesRaw;

    // ------- per-wave constants for phase 1 (live across all 256 steps) -------
    int wv   = threadIdx.x >> 5;
    int lane = threadIdx.x & 31;
    int wave = (blockIdx.x << 3) + wv;        // 0..1023
    int mt   = wave & 3;
    int np   = wave >> 2;
    int n0   = np << 5;
    int cell = n0 >> 12;                      // 0 = F, 1 = R
    int ncol = n0 & 4095;

    const __bf16* Wcat = (const __bf16*)(cell ? WcatRRaw : WcatFRaw);
    int m0    = mt << 4;
    int row_a = m0 + (lane & 15);
    int ksel  = (lane >> 4) << 4;
    int nc0   = ncol + (lane & 15);
    int nc1   = nc0 + 16;

    const __bf16* b0 = Wcat + (size_t)nc0 * KCAT + ksel;
    const __bf16* b1 = Wcat + (size_t)nc1 * KCAT + ksel;
    const __bf16* aXbase = Xbf + (size_t)row_a * SEQ * INP + ksel;
    const __bf16* aT = Tbf + (size_t)row_a * HID + ksel;
    const __bf16* aOnesP = ones + ksel;

    float*  cbuf = cell ? cR : cF;
    __bf16* hbuf = (__bf16*)(cell ? hRRaw : hFRaw);
    int jbase = ncol >> 2;

#pragma clang loop unroll(disable)
    for (int t = 0; t < SEQ; ++t) {
        OPT_BARRIER();                        // no load hoisting across steps

        // =============== phase 1: gate GEMM + LSTM activations ===============
        v8f acc0 = {}, acc1 = {};

        const __bf16* aX = aXbase + (size_t)t * INP;
        for (int k = 0; k < INP; k += 32) {
            v16bf a  = *(const v16bf*)(aX + k);
            v16bf w0 = *(const v16bf*)(b0 + k);
            v16bf w1 = *(const v16bf*)(b1 + k);
            acc0 = __builtin_amdgcn_wmma_f32_16x16x32_bf16(false, a, false, w0, (short)0, acc0, false, false);
            acc1 = __builtin_amdgcn_wmma_f32_16x16x32_bf16(false, a, false, w1, (short)0, acc1, false, false);
        }
        for (int k = 0; k < HID; k += 32) {
            v16bf a  = *(const v16bf*)(aT + k);
            v16bf w0 = *(const v16bf*)(b0 + INP + k);
            v16bf w1 = *(const v16bf*)(b1 + INP + k);
            acc0 = __builtin_amdgcn_wmma_f32_16x16x32_bf16(false, a, false, w0, (short)0, acc0, false, false);
            acc1 = __builtin_amdgcn_wmma_f32_16x16x32_bf16(false, a, false, w1, (short)0, acc1, false, false);
        }
        {   // bias tile: constant-1 A column x (bih+bhh) weight column
            v16bf a  = *(const v16bf*)aOnesP;
            v16bf w0 = *(const v16bf*)(b0 + INP + HID);
            v16bf w1 = *(const v16bf*)(b1 + INP + HID);
            acc0 = __builtin_amdgcn_wmma_f32_16x16x32_bf16(false, a, false, w0, (short)0, acc0, false, false);
            acc1 = __builtin_amdgcn_wmma_f32_16x16x32_bf16(false, a, false, w1, (short)0, acc1, false, false);
        }

        // regroup through LDS: C/D layout -> per-lane (i,f,g,o) quads
#pragma unroll
        for (int v = 0; v < 8; ++v) {
            int r = ((lane >> 4) << 3) + v;
            ltile[wv][r][lane & 15]        = acc0[v];
            ltile[wv][r][(lane & 15) + 16] = acc1[v];
        }
        __syncthreads();

        for (int it = 0; it < 4; ++it) {
            int q = lane + (it << 5);
            int r = q >> 3;
            int u = q & 7;
            float iv = ltile[wv][r][4 * u + 0];
            float fv = ltile[wv][r][4 * u + 1];
            float gv = ltile[wv][r][4 * u + 2];
            float ov = ltile[wv][r][4 * u + 3];
            int b = m0 + r;
            int j = jbase + u;
            float* cp   = cbuf + (size_t)b * HID + j;
            float  cold = *cp;
            float si = 1.0f / (1.0f + __expf(-iv));
            float sf = 1.0f / (1.0f + __expf(-fv));
            float so = 1.0f / (1.0f + __expf(-ov));
            float c2 = sf * cold + si * tanhf(gv);
            *cp = c2;
            hbuf[(size_t)b * HID + j] = (__bf16)(so * tanhf(c2));
        }

        grid_sync(bar_cnt, bar_gen);          // h complete, device-visible
        OPT_BARRIER();

        // ========= phase 2: attention + softmax + outer (blocks 0..3) =========
        if (blockIdx.x < 4) {
            int bm0 = blockIdx.x << 4;        // 16-batch tile of this block

            for (int nt = wv; nt < 14; nt += 8) {
                bool isR = (nt >= 10);
                int  an0 = (isR ? nt - 10 : nt) << 4;
                const __bf16* A = (const __bf16*)(isR ? hRRaw : hFRaw);
                const __bf16* W = (const __bf16*)(isR ? WaRRaw : WaFRaw);
                const __bf16* ap = A + (size_t)(bm0 + (lane & 15)) * HID + ksel;
                const __bf16* bp = W + (size_t)(an0 + (lane & 15)) * HID + ksel;
                v8f acc = {};
                for (int k = 0; k < HID; k += 32) {
                    v16bf a = *(const v16bf*)(ap + k);
                    v16bf w = *(const v16bf*)(bp + k);
                    acc = __builtin_amdgcn_wmma_f32_16x16x32_bf16(false, a, false, w, (short)0, acc, false, false);
                }
                int nn = an0 + (lane & 15);
#pragma unroll
                for (int v = 0; v < 8; ++v) {
                    int r = ((lane >> 4) << 3) + v;
                    if (isR) lgR_s[r][nn] = acc[v];
                    else     lgF_s[r][nn] = acc[v];
                }
            }
            __syncthreads();

            int tid = threadIdx.x;
            if (tid < 16) {                   // softmax over F logits (150)
                int r = tid, b = bm0 + r;
                float mx = -3.4e38f;
                for (int k = 0; k < NS; ++k) {
                    float v = lgF_s[r][k] + WaFb[k];
                    lgF_s[r][k] = v;
                    mx = fmaxf(mx, v);
                }
                float sum = 0.0f;
                for (int k = 0; k < NS; ++k) {
                    float e = __expf(lgF_s[r][k] - mx);
                    lgF_s[r][k] = e;
                    sum += e;
                }
                float inv = 1.0f / sum;
                for (int k = 0; k < NS; ++k) {
                    float a = lgF_s[r][k] * inv;
                    lgF_s[r][k] = a;
                    outAF[((size_t)b * SEQ + t) * NS + k] = a;
                }
            } else if (tid < 32) {            // softmax over R logits (50)
                int r = tid - 16, b = bm0 + r;
                float mx = -3.4e38f;
                for (int k = 0; k < NR; ++k) {
                    float v = lgR_s[r][k] + WaRb[k];
                    lgR_s[r][k] = v;
                    mx = fmaxf(mx, v);
                }
                float sum = 0.0f;
                for (int k = 0; k < NR; ++k) {
                    float e = __expf(lgR_s[r][k] - mx);
                    lgR_s[r][k] = e;
                    sum += e;
                }
                float inv = 1.0f / sum;
                for (int k = 0; k < NR; ++k) {
                    float a = lgR_s[r][k] * inv;
                    lgR_s[r][k] = a;
                    outAR[((size_t)b * SEQ + t) * NR + k] = a;
                }
            }
            __syncthreads();

            for (int idx = tid; idx < 512; idx += 256) {   // itemF
                int r = idx >> 5, u = idx & 31;
                float s = Fb[u];
                for (int k = 0; k < NS; ++k) s += lgF_s[r][k] * Fw[u * NS + k];
                itF_s[r][u] = s;
            }
            for (int idx = tid; idx < 512; idx += 256) {   // itemR
                int r = idx >> 5, u = idx & 31;
                float s = Rb[u];
                for (int k = 0; k < NR; ++k) s += lgR_s[r][k] * Rw[u * NR + k];
                itR_s[r][u] = s;
            }
            __syncthreads();

            __bf16* Tw = (__bf16*)TbfRaw;     // outer product -> next T
            for (int idx = tid; idx < 16 * HID; idx += 256) {
                int r = idx >> 10, i = idx & 1023;
                float val = itF_s[r][i >> 5] * itR_s[r][i & 31];
                int b = bm0 + r;
                outT[((size_t)b * SEQ + t) * HID + i] = val;
                Tw[(size_t)b * HID + i] = (__bf16)val;
            }
        }

        grid_sync(bar_cnt, bar_gen);          // T_t complete, device-visible
    }
}

// ===================================================================
// host: workspace layout + launch sequence
// ===================================================================
extern "C" void kernel_launch(void* const* d_in, const int* in_sizes, int n_in,
                              void* d_out, int out_size, void* d_ws, size_t ws_size,
                              hipStream_t stream)
{
    const float* x    = (const float*)d_in[0];
    const float* WihF = (const float*)d_in[1];
    const float* WhhF = (const float*)d_in[2];
    const float* bihF = (const float*)d_in[3];
    const float* bhhF = (const float*)d_in[4];
    const float* WihR = (const float*)d_in[5];
    const float* WhhR = (const float*)d_in[6];
    const float* bihR = (const float*)d_in[7];
    const float* bhhR = (const float*)d_in[8];
    const float* WaFw = (const float*)d_in[9];
    const float* WaFb = (const float*)d_in[10];
    const float* WaRw = (const float*)d_in[11];
    const float* WaRb = (const float*)d_in[12];
    const float* Fw   = (const float*)d_in[13];
    const float* Fb   = (const float*)d_in[14];
    const float* Rw   = (const float*)d_in[15];
    const float* Rb   = (const float*)d_in[16];

    float* outT  = (float*)d_out;                         // [B,S,H]
    float* outAF = outT + (size_t)BATCH * SEQ * HID;      // [B,S,NS]
    float* outAR = outAF + (size_t)BATCH * SEQ * NS;      // [B,S,NR]

    // ---- workspace layout (all section sizes are multiples of 256 B) ----
    char* w = (char*)d_ws;
    unsigned short* XBF   = (unsigned short*)w; w += (size_t)BATCH * SEQ * INP * 2; // 13.6 MB
    unsigned short* WCATF = (unsigned short*)w; w += (size_t)H4 * KCAT * 2;         // 12.1 MB
    unsigned short* WCATR = (unsigned short*)w; w += (size_t)H4 * KCAT * 2;         // 12.1 MB
    unsigned short* WAF   = (unsigned short*)w; w += (size_t)NSP * HID * 2;
    unsigned short* WAR   = (unsigned short*)w; w += (size_t)NRP * HID * 2;
    unsigned short* HF    = (unsigned short*)w; w += (size_t)BATCH * HID * 2;
    unsigned short* HR    = (unsigned short*)w; w += (size_t)BATCH * HID * 2;
    unsigned short* TBF   = (unsigned short*)w; w += (size_t)BATCH * HID * 2;
    float*          CF    = (float*)w;          w += (size_t)BATCH * HID * 4;
    float*          CR    = (float*)w;          w += (size_t)BATCH * HID * 4;
    unsigned short* ONES  = (unsigned short*)w; w += 256;
    int*            BAR   = (int*)w;            w += 256;

    // ---- one-time (per launch) bf16 staging: x, combined gate weights, Wa ----
    cvt_pad_kernel<<<4096, 256, 0, stream>>>(x, XBF, BATCH * SEQ, INDIM, BATCH * SEQ, INP);
    build_wcat_kernel<<<4096, 256, 0, stream>>>(WihF, WhhF, bihF, bhhF, WCATF);
    build_wcat_kernel<<<4096, 256, 0, stream>>>(WihR, WhhR, bihR, bhhR, WCATR);
    cvt_pad_kernel<<<256, 256, 0, stream>>>(WaFw, WAF, NS, HID, NSP, HID);
    cvt_pad_kernel<<<128, 256, 0, stream>>>(WaRw, WAR, NR, HID, NRP, HID);
    init_state_kernel<<<256, 256, 0, stream>>>(TBF, CF, CR, ONES, BAR);

    // ---- the whole 256-step scan: ONE persistent kernel ----
    tpr_scan_kernel<<<NBLK, 256, 0, stream>>>(XBF, TBF, WCATF, WCATR, WAF, WAR, ONES,
                                              CF, CR, HF, HR,
                                              WaFb, WaRb, Fw, Fb, Rw, Rb,
                                              outT, outAF, outAR, BAR);
    (void)in_sizes; (void)n_in; (void)out_size; (void)ws_size;
}
